// TriU_30640296690077
// MI455X (gfx1250) — matrix operand order repaired and checked by reading, not compiled, not run on hardware
//
#include <hip/hip_runtime.h>
#include <stdint.h>

// ---------------------------------------------------------------------------
// TriU prefix-linear == masked GEMM: out[m, i] = sum_{k<=i} X[m,k] * W[i,k] + b[i]
// M = 64*1024 = 65536, K = N = 512.
// Strategy: bf16 split-precision WMMA (Xh*Wh + Xh*Wl + Xl*Wh) on
// v_wmma_f32_16x16x32_bf16, W pre-masked/transposed/split/swizzled into WMMA
// B-fragment order in d_ws (2 MB, L2-resident), X staged through LDS in
// A-fragment order. Workgroup tile: 128 rows x 256 cols, 8 wave32 waves,
// each wave 4 m-tiles x 4 n-tiles (64x64).
// ---------------------------------------------------------------------------

typedef __attribute__((ext_vector_type(16))) __bf16 v16bf;
typedef __attribute__((ext_vector_type(8)))  float  v8f;

union Frag {
  v16bf bf;
  uint4 q[2];
};

__device__ __forceinline__ uint32_t f2bf(float f) {           // fp32 -> bf16 (RNE)
  uint32_t u = __float_as_uint(f);
  return (u + 0x7FFFu + ((u >> 16) & 1u)) >> 16;
}
__device__ __forceinline__ float bf2f(uint32_t h) { return __uint_as_float(h << 16); }

// ---------------------------------------------------------------------------
// Prep: build masked, transposed, split W in WMMA B-fragment order.
// Block (nt, kt) = 32x16 tile of B[k][n] (k = kt*32.., n = nt*16..), stored as
// 2048 B: lane l (0..31) gets 32 B at l*32; reg r (0..7) dword at r*4 holds
// K = kt*32 + 2r + 16*(l>>4) (+1 in high half), column n = nt*16 + (l&15).
// Block index = nt*16 + kt.  hi array then lo array (1 MB each).
// ---------------------------------------------------------------------------
__global__ __launch_bounds__(256) void TriU_prepW(const float* __restrict__ W,
                                                  uint32_t* __restrict__ hi,
                                                  uint32_t* __restrict__ lo) {
  int g   = blockIdx.x * 256 + threadIdx.x;      // 0 .. 131071
  int r   = g & 7;
  int l   = (g >> 3) & 31;
  int blk = g >> 8;                              // 0 .. 511  == nt*16 + kt
  int kt  = blk & 15;
  int n   = (blk >> 4) * 16 + (l & 15);
  int k0  = kt * 32 + 2 * r + 16 * (l >> 4);
  float v0 = (k0     <= n) ? W[n * 512 + k0]     : 0.0f;   // tril mask: keep k <= n
  float v1 = (k0 + 1 <= n) ? W[n * 512 + k0 + 1] : 0.0f;
  uint32_t h0 = f2bf(v0),               h1 = f2bf(v1);
  uint32_t l0 = f2bf(v0 - bf2f(h0)),    l1 = f2bf(v1 - bf2f(h1));
  int idx = blk * 512 + l * 8 + r;               // dword index inside hi/lo array
  hi[idx] = h0 | (h1 << 16);
  lo[idx] = l0 | (l1 << 16);
}

// ---------------------------------------------------------------------------
// Main GEMM.
// grid = (M/128, 512/256) = (512, 2), block = 256 threads = 8 wave32 waves.
// wave w: wave_m = w>>2 (m-tiles 4*wave_m..+3), wave_n = w&3 (n-tiles 4*wave_n..+3).
// ---------------------------------------------------------------------------
__global__ __launch_bounds__(256) void TriU_gemm(const float* __restrict__ X,
                                                 const uint4* __restrict__ Bhi,
                                                 const uint4* __restrict__ Blo,
                                                 const float* __restrict__ bias,
                                                 float* __restrict__ out) {
  __shared__ uint32_t lds[4096];   // 16 KB: A fragments, 8 m-tiles x {hi,lo} x 1 KB

  const int tid    = threadIdx.x;
  const int lane   = tid & 31;
  const int w      = tid >> 5;
  const int wave_m = w >> 2;                 // 0..1
  const int wave_n = w & 3;                  // 0..3
  const int m_base = blockIdx.x * 128;
  const int nt_wg  = blockIdx.y * 16;        // first n-tile of this WG

  // loader-side constants: thread i covers row i>>1, k-half (i&1)*16
  const int lrow = tid >> 1;                 // 0..127
  const int t_l  = lrow >> 4;                // m-tile of loaded row
  const int lr15 = lrow & 15;
  const int kh   = (tid & 1) * 16;

  v8f acc[4][4];
  #pragma unroll
  for (int a = 0; a < 4; ++a)
    #pragma unroll
    for (int b = 0; b < 4; ++b) acc[a][b] = {};

  for (int kt = 0; kt < 16; ++kt) {
    __syncthreads();
    // ---- stage A: 128 rows x 32 K fp32 -> bf16 hi/lo fragments in LDS ----
    uint2* lds2 = (uint2*)lds;
    #pragma unroll
    for (int q = 0; q < 4; ++q) {
      int kl = kh + 4 * q;                                   // 0..28, mult of 4
      float4 x = *(const float4*)(X + (size_t)(m_base + lrow) * 512 + kt * 32 + kl);
      // contiguous A layout (matches B): khalf=k>>4, r=(k&15)>>1
      int khalf = kl >> 4;
      int lA    = lr15 + 16 * khalf;
      int r0    = (kl & 15) >> 1;                            // even
      uint32_t h0 = f2bf(x.x), h1 = f2bf(x.y), h2 = f2bf(x.z), h3 = f2bf(x.w);
      uint32_t e0 = f2bf(x.x - bf2f(h0)), e1 = f2bf(x.y - bf2f(h1));
      uint32_t e2 = f2bf(x.z - bf2f(h2)), e3 = f2bf(x.w - bf2f(h3));
      int bi = (t_l * 2) * 128 + lA * 4 + (r0 >> 1);         // uint2 index (hi part)
      lds2[bi]       = make_uint2(h0 | (h1 << 16), h2 | (h3 << 16));
      lds2[bi + 128] = make_uint2(e0 | (e1 << 16), e2 | (e3 << 16));  // lo part
    }
    __syncthreads();

    // ---- compute: load A frags (LDS) once, B frags (global/L2) per n-tile ----
    const uint4* ldsq = (const uint4*)lds;
    Frag ah[4], al[4];
    #pragma unroll
    for (int t2 = 0; t2 < 4; ++t2) {
      int t  = wave_m * 4 + t2;
      int bq = t * 128 + lane * 2;                           // uint4 index
      ah[t2].q[0] = ldsq[bq];      ah[t2].q[1] = ldsq[bq + 1];
      al[t2].q[0] = ldsq[bq + 64]; al[t2].q[1] = ldsq[bq + 65];
    }
    #pragma unroll
    for (int j = 0; j < 4; ++j) {
      int ntg = nt_wg + wave_n * 4 + j;
      size_t boff = (((size_t)(ntg * 16 + kt) * 2048) + (size_t)lane * 32) >> 4;
      Frag bh, bl;
      bh.q[0] = Bhi[boff]; bh.q[1] = Bhi[boff + 1];
      bl.q[0] = Blo[boff]; bl.q[1] = Blo[boff + 1];
      if (kt < 15) {                                         // next-K prefetch (L2)
        __builtin_prefetch(&Bhi[boff + 128], 0, 1);
        __builtin_prefetch(&Blo[boff + 128], 0, 1);
      }
      #pragma unroll
      for (int t2 = 0; t2 < 4; ++t2) {
        acc[t2][j] = __builtin_amdgcn_wmma_f32_16x16x32_bf16(
            false, ah[t2].bf, false, bh.bf, (short)0, acc[t2][j], false, false);
        acc[t2][j] = __builtin_amdgcn_wmma_f32_16x16x32_bf16(
            false, ah[t2].bf, false, bl.bf, (short)0, acc[t2][j], false, false);
        acc[t2][j] = __builtin_amdgcn_wmma_f32_16x16x32_bf16(
            false, al[t2].bf, false, bh.bf, (short)0, acc[t2][j], false, false);
      }
    }
  }

  // ---- epilogue: bias + store (C/D layout: VGPR v -> row v + 8*(lane>=16)) ----
  const int lhi = lane >> 4, lln = lane & 15;
  #pragma unroll
  for (int j = 0; j < 4; ++j) {
    int col  = (nt_wg + wave_n * 4 + j) * 16 + lln;
    float bv = bias[col];
    #pragma unroll
    for (int t2 = 0; t2 < 4; ++t2) {
      int row0 = m_base + (wave_m * 4 + t2) * 16 + 8 * lhi;
      #pragma unroll
      for (int v = 0; v < 8; ++v)
        out[(size_t)(row0 + v) * 512 + col] = acc[t2][j][v] + bv;
    }
  }
}

// ---------------------------------------------------------------------------
extern "C" void kernel_launch(void* const* d_in, const int* in_sizes, int n_in,
                              void* d_out, int out_size, void* d_ws, size_t ws_size,
                              hipStream_t stream) {
  (void)in_sizes; (void)n_in; (void)out_size; (void)ws_size;
  const float* X    = (const float*)d_in[0];   // (64,1024,512) fp32
  const float* W    = (const float*)d_in[1];   // (512,512) fp32
  const float* bias = (const float*)d_in[2];   // (512,) fp32
  float* out        = (float*)d_out;

  // workspace: packed W fragments, hi then lo (1 MB each; needs 2 MB of d_ws)
  uint32_t* hi = (uint32_t*)d_ws;
  uint32_t* lo = hi + (1u << 18);

  TriU_prepW<<<512, 256, 0, stream>>>(W, hi, lo);

  dim3 grid(512, 2);   // 65536/128 row tiles, 512/256 col tiles
  TriU_gemm<<<grid, 256, 0, stream>>>(X, (const uint4*)hi, (const uint4*)lo, bias, out);
}
